// RepulsionLoss_41463614276098
// MI455X (gfx1250) — compile-verified, matrix operand order checked by reading, not run on hardware
//
#include <hip/hip_runtime.h>
#include <hip/hip_bf16.h>

typedef __attribute__((ext_vector_type(16))) _Float16 v16h;
typedef __attribute__((ext_vector_type(8)))  _Float16 v8h;
typedef __attribute__((ext_vector_type(4)))  _Float16 v4h;
typedef __attribute__((ext_vector_type(8)))  float    v8f;

// Problem constants (fixed by the reference)
constexpr int   N        = 2048;
constexpr int   B        = 16;
constexpr float STRENGTH = 0.1f;
constexpr float RADIUS   = 2310.0f;

constexpr int J_PER_WAVE  = 16;                  // one 16x16 WMMA tile of j per wave
constexpr int WAVES       = 8;                   // wave32: 256 threads = 8 waves
constexpr int J_PER_BLOCK = J_PER_WAVE * WAVES;  // 128
constexpr int THREADS     = WAVES * 32;          // 256
constexpr int HALF_N      = N / 2;               // i-split for the two WMMA chains

// --- zero the S0/S1 accumulators in workspace (graph-capture safe) ----------
__global__ void rep_init(float* __restrict__ ws) {
  if (threadIdx.x < 2 * B) ws[threadIdx.x] = 0.0f;
}

// --- O(B*N^2) phase: pairwise L1 distances via WMMA row-sum reduction -------
// Block = (b, 128-wide j chunk). x[b,:] staged in LDS in f32 (weights) and f16
// (WMMA feed). Each wave owns a 16-row j tile and runs TWO independent WMMA
// accumulation chains over disjoint halves of the i range; the independent
// VALU/load streams fill the 4-coexec WMMA->VALU hazard slots and hide LDS
// latency. dist[j] accumulates in f32 inside the WMMA C/D operands.
__global__ __launch_bounds__(THREADS)
void rep_dist(const float* __restrict__ x, float* __restrict__ ws) {
  __shared__ __align__(16) float    xs[N];   // f32 copy (weight phase)
  __shared__ __align__(16) _Float16 xh[N];   // f16 copy (WMMA feed)
  __shared__ float dist_s[J_PER_BLOCK];
  __shared__ float red0[J_PER_BLOCK];
  __shared__ float red1[J_PER_BLOCK];

  const int b     = blockIdx.y;
  const int jbase = blockIdx.x * J_PER_BLOCK;
  const int tid   = threadIdx.x;
  const int lane  = tid & 31;
  const int wave  = tid >> 5;

  // Stage x[b,:] into LDS (f32 and f16) -- 2 float4 per thread
  const float4* xg  = reinterpret_cast<const float4*>(x + b * N);
  float4*       xs4 = reinterpret_cast<float4*>(xs);
#pragma unroll
  for (int t = 0; t < (N / 4) / THREADS; ++t) {
    const int idx = tid + t * THREADS;       // float4 index
    float4 u = xg[idx];
    xs4[idx] = u;
    v4h hv;
    hv[0] = (_Float16)u.x; hv[1] = (_Float16)u.y;
    hv[2] = (_Float16)u.z; hv[3] = (_Float16)u.w;
    *reinterpret_cast<v4h*>(&xh[idx * 4]) = hv;
  }
  __syncthreads();

  // A-matrix 16x32 f16 layout (ISA 7.12.2): lane L<16 -> row M=L, K {0..7,16..23};
  // lane L+16 -> same row, K {8..15,24..31}.
  const int      m    = lane & 15;
  const _Float16 xjh  = xh[jbase + wave * J_PER_WAVE + m];
  const int      koff = (lane < 16) ? 0 : 8;

  v8h xjv;
#pragma unroll
  for (int t = 0; t < 8; ++t) xjv[t] = xjh;

  v16h ones;
#pragma unroll
  for (int t = 0; t < 16; ++t) ones[t] = (_Float16)1.0f;

  v8f c0 = {};  // chain 0: i in [0, N/2)
  v8f c1 = {};  // chain 1: i in [N/2, N)

  for (int ibase = 0; ibase < HALF_N; ibase += 32) {
    // chain 0 tile
    v8h g0 = *reinterpret_cast<const v8h*>(&xh[ibase + koff]);
    v8h g1 = *reinterpret_cast<const v8h*>(&xh[ibase + koff + 16]);
    // chain 1 tile
    v8h h0 = *reinterpret_cast<const v8h*>(&xh[HALF_N + ibase + koff]);
    v8h h1 = *reinterpret_cast<const v8h*>(&xh[HALF_N + ibase + koff + 16]);

    v8h d0 = __builtin_elementwise_abs(g0 - xjv);  // v_pk sub + f16 abs (and)
    v8h d1 = __builtin_elementwise_abs(g1 - xjv);
    v8h e0 = __builtin_elementwise_abs(h0 - xjv);
    v8h e1 = __builtin_elementwise_abs(h1 - xjv);

    v16h a0 = __builtin_shufflevector(d0, d1, 0, 1, 2, 3, 4, 5, 6, 7,
                                      8, 9, 10, 11, 12, 13, 14, 15);
    v16h a1 = __builtin_shufflevector(e0, e1, 0, 1, 2, 3, 4, 5, 6, 7,
                                      8, 9, 10, 11, 12, 13, 14, 15);

    // D[m,n] = sum_k A[m,k] * 1 + C[m,n]  (every column identical = row sums)
    c0 = __builtin_amdgcn_wmma_f32_16x16x32_f16(false, a0, false, ones,
                                                (short)0, c0, false, false);
    c1 = __builtin_amdgcn_wmma_f32_16x16x32_f16(false, a1, false, ones,
                                                (short)0, c1, false, false);
  }

  v8f c = c0 + c1;  // merge the two i-halves (f32)

  // Column 0 of D: lane 0 holds M=0..7 in c[0..7], lane 16 holds M=8..15.
  if (lane == 0 || lane == 16) {
    const int mrow = (lane == 0) ? 0 : 8;
#pragma unroll
    for (int r = 0; r < 8; ++r)
      dist_s[wave * J_PER_WAVE + mrow + r] = c[r];
  }
  __syncthreads();

  // Threshold -> weights; block-partial S0 = sum w, S1 = sum w*x
  if (tid < J_PER_BLOCK) {
    const float d = dist_s[tid];
    const float w = (d < RADIUS) ? STRENGTH : 0.0f;
    red0[tid] = w;
    red1[tid] = w * xs[jbase + tid];
  }
  __syncthreads();
  for (int s = J_PER_BLOCK / 2; s > 0; s >>= 1) {
    if (tid < s) { red0[tid] += red0[tid + s]; red1[tid] += red1[tid + s]; }
    __syncthreads();
  }
  if (tid == 0) {
    atomicAdd(&ws[2 * b + 0], red0[0]);  // S0[b]
    atomicAdd(&ws[2 * b + 1], red1[0]);  // S1[b]
  }
}

// --- O(B*N) phase: loss = mean_b sqrt( sum_j (S1[b] - x[b,j]*S0[b])^2 ) -----
__global__ __launch_bounds__(256)
void rep_finalize(const float* __restrict__ x, const float* __restrict__ ws,
                  float* __restrict__ out) {
  __shared__ float red[256];
  const int tid = threadIdx.x;
  float acc = 0.0f;
  for (int b = 0; b < B; ++b) {
    const float S0 = ws[2 * b + 0];
    const float S1 = ws[2 * b + 1];
    float local = 0.0f;
    for (int j = tid; j < N; j += 256) {
      const float t = S1 - x[b * N + j] * S0;
      local += t * t;
    }
    red[tid] = local;
    __syncthreads();
    for (int s = 128; s > 0; s >>= 1) {
      if (tid < s) red[tid] += red[tid + s];
      __syncthreads();
    }
    if (tid == 0) acc += sqrtf(red[0]);
    __syncthreads();
  }
  if (tid == 0) out[0] = acc / (float)B;
}

extern "C" void kernel_launch(void* const* d_in, const int* in_sizes, int n_in,
                              void* d_out, int out_size, void* d_ws, size_t ws_size,
                              hipStream_t stream) {
  const float* x   = (const float*)d_in[0];
  float*       ws  = (float*)d_ws;   // 2*B floats: S0/S1 per batch
  float*       out = (float*)d_out;  // scalar loss

  rep_init<<<1, 64, 0, stream>>>(ws);
  dim3 grid(N / J_PER_BLOCK, B);  // (16, 16) = 256 blocks
  rep_dist<<<grid, THREADS, 0, stream>>>(x, ws);
  rep_finalize<<<1, 256, 0, stream>>>(x, ws, out);
}
